// ExpertNetworkOptimized_75256416961126
// MI455X (gfx1250) — compile-verified, matrix operand order checked by reading, not run on hardware
//
#include <hip/hip_runtime.h>
#include <hip/hip_bf16.h>

typedef __attribute__((ext_vector_type(16))) __bf16 v16bf;
typedef __attribute__((ext_vector_type(8)))  float  v8f;
typedef __attribute__((ext_vector_type(4)))  float  f32x4;

#define NTOK   32768
#define DIM    512      // IN_DIM == HID == 512
#define NEXP   8
#define ROWT   128      // rows per tile
#define MAXTILES 320    // >= NTOK/ROWT + NEXP
#define EPSBN  1e-5f

// ---------------- workspace layout (ints first, bf16 weights after) ----------
// ws_i[0..7]    counts
// ws_i[8..15]   cursors
// ws_i[16]      n_tiles
// ws_i[32..]    tile_expert  [MAXTILES]
// ws_i[352..]   tile_rowbase [MAXTILES]
// ws_i[672..]   tile_rows    [MAXTILES]
// ws_i[1024..]  perm         [NTOK]
// byte offset 262144: Whi (8*512*512 bf16 = 4 MB), then Wlo (4 MB)
#define OFF_COUNT 0
#define OFF_CURS  8
#define OFF_NT    16
#define OFF_TE    32
#define OFF_TRB   352
#define OFF_TNR   672
#define OFF_PERM  1024
#define OFF_WBF   262144  // bytes

__global__ void zero_meta_k(int* ws_i) {
    if (threadIdx.x < 16) ws_i[OFF_COUNT + threadIdx.x] = 0;
}

__global__ void hist_k(const int* __restrict__ eidx, int* __restrict__ ws_i) {
    int t = blockIdx.x * 256 + threadIdx.x;
    if (t < NTOK) atomicAdd(&ws_i[OFF_COUNT + eidx[t]], 1);
}

__global__ void build_tiles_k(int* __restrict__ ws_i) {
    if (threadIdx.x != 0 || blockIdx.x != 0) return;
    int off = 0, nt = 0;
    for (int e = 0; e < NEXP; ++e) {
        int c = ws_i[OFF_COUNT + e];
        ws_i[OFF_CURS + e] = off;
        int nt_e = (c + ROWT - 1) / ROWT;
        for (int i = 0; i < nt_e; ++i) {
            ws_i[OFF_TE  + nt] = e;
            ws_i[OFF_TRB + nt] = off + i * ROWT;
            int rem = c - i * ROWT;
            ws_i[OFF_TNR + nt] = rem < ROWT ? rem : ROWT;
            ++nt;
        }
        off += c;
    }
    ws_i[OFF_NT] = nt;
}

__global__ void scatter_k(const int* __restrict__ eidx, int* __restrict__ ws_i) {
    int t = blockIdx.x * 256 + threadIdx.x;
    if (t < NTOK) {
        int e = eidx[t];
        int p = atomicAdd(&ws_i[OFF_CURS + e], 1);
        ws_i[OFF_PERM + p] = t;
    }
}

// Convert W[e][K][N] (fp32) into bf16 hi/lo fragments pre-swizzled for WMMA B.
// Fragment (e, ct, kt): lane n -> column N = ct*16 + (n&15); hf = n>>4;
// element j: K = kt*32 + (j<8 ? hf*8+j : 16+hf*8+(j-8))   [16-bit 32xK layout]
__global__ void convw_k(const float* __restrict__ W, __bf16* __restrict__ Whi,
                        __bf16* __restrict__ Wlo) {
    int t = blockIdx.x * 256 + threadIdx.x;     // 131072 threads total
    int lane = t & 31;
    int kt   = (t >> 5) & 15;
    int ct   = (t >> 9) & 31;
    int e    = t >> 14;
    int hf   = lane >> 4;
    int n    = ct * 16 + (lane & 15);
    const float* wp = W + (size_t)e * DIM * DIM;
    v16bf hi, lo;
#pragma unroll
    for (int j = 0; j < 16; ++j) {
        int K = kt * 32 + (j < 8 ? hf * 8 + j : 16 + hf * 8 + (j - 8));
        float f = wp[(size_t)K * DIM + n];
        __bf16 h = (__bf16)f;
        hi[j] = h;
        lo[j] = (__bf16)(f - (float)h);
    }
    size_t base = (size_t)t * 16;
    *(v16bf*)(Whi + base) = hi;
    *(v16bf*)(Wlo + base) = lo;
}

// Grouped GEMM + bias + BN(inference) + SiLU.
// Block: 256 threads = 8 waves. Tile: 128 rows x 64 cols; wave w owns rows
// [16w,16w+16) x all 64 cols (4 col-subtiles of 16).
__global__ __launch_bounds__(256)
void moe_gemm_k(const float* __restrict__ x,
                const float* __restrict__ biases,
                const float* __restrict__ bn_w, const float* __restrict__ bn_b,
                const float* __restrict__ rmean, const float* __restrict__ rvar,
                const int* __restrict__ ws_i,
                const __bf16* __restrict__ Whi, const __bf16* __restrict__ Wlo,
                float* __restrict__ out) {
    int tile = blockIdx.x;
    if (tile >= ws_i[OFF_NT]) return;               // uniform exit
    const int e       = ws_i[OFF_TE  + tile];
    const int rowbase = ws_i[OFF_TRB + tile];
    const int nrows   = ws_i[OFF_TNR + tile];
    const int colblk  = blockIdx.y;                 // 0..7 -> 64 cols each
    const int wave    = threadIdx.x >> 5;
    const int lane    = threadIdx.x & 31;
    const int hf      = lane >> 4;
    const int* perm   = ws_i + OFF_PERM;

    // A row for this lane (clamped for padded rows; masked at store)
    int r_local = wave * 16 + (lane & 15);
    int r_clamp = r_local < nrows ? r_local : (nrows - 1);
    int grow_a  = perm[rowbase + r_clamp];
    const float* ap = x + (size_t)grow_a * DIM;

    v8f acc[4];
#pragma unroll
    for (int c = 0; c < 4; ++c) acc[c] = (v8f)0.0f;

    for (int kt = 0; kt < DIM / 32; ++kt) {
        const int k0 = kt * 32;
        // ---- A fragment: two 8-float runs per lane, split into bf16 hi/lo
        f32x4 a0 = *(const f32x4*)(ap + k0 + hf * 8);
        f32x4 a1 = *(const f32x4*)(ap + k0 + hf * 8 + 4);
        f32x4 a2 = *(const f32x4*)(ap + k0 + 16 + hf * 8);
        f32x4 a3 = *(const f32x4*)(ap + k0 + 16 + hf * 8 + 4);
        float fa[16];
#pragma unroll
        for (int j = 0; j < 4; ++j) {
            fa[j] = a0[j]; fa[4 + j] = a1[j]; fa[8 + j] = a2[j]; fa[12 + j] = a3[j];
        }
        v16bf ahi, alo;
#pragma unroll
        for (int j = 0; j < 16; ++j) {
            __bf16 h = (__bf16)fa[j];
            ahi[j] = h;
            alo[j] = (__bf16)(fa[j] - (float)h);
        }
        // ---- 4 col-subtiles: B fragments are preswizzled, coalesced loads
#pragma unroll
        for (int c = 0; c < 4; ++c) {
            int ctg = colblk * 4 + c;
            size_t fb = ((((size_t)e * 32 + ctg) * 16 + kt) * 32 + lane) * 16;
            v16bf bhi = *(const v16bf*)(Whi + fb);
            v16bf blo = *(const v16bf*)(Wlo + fb);
            acc[c] = __builtin_amdgcn_wmma_f32_16x16x32_bf16(
                false, ahi, false, bhi, (short)0, acc[c], false, false);
            acc[c] = __builtin_amdgcn_wmma_f32_16x16x32_bf16(
                false, alo, false, bhi, (short)0, acc[c], false, false);
            acc[c] = __builtin_amdgcn_wmma_f32_16x16x32_bf16(
                false, ahi, false, blo, (short)0, acc[c], false, false);
        }
    }

    // ---- epilogue: bias + BN + SiLU, scatter per C-matrix layout
#pragma unroll
    for (int c = 0; c < 4; ++c) {
        int col = colblk * 64 + c * 16 + (lane & 15);
        int pe  = e * DIM + col;
        float bias  = biases[pe];
        float mean  = rmean[pe];
        float var   = rvar[pe];
        float gamma = bn_w[pe];
        float beta  = bn_b[pe];
        float scale = rsqrtf(var + EPSBN) * gamma;
        float shift = beta - mean * scale;
#pragma unroll
        for (int j = 0; j < 8; ++j) {
            int rl = wave * 16 + j + hf * 8;   // C VGPR j: M=j (lanes 0-15) / j+8
            if (rl < nrows) {
                int g = perm[rowbase + rl];
                float y = (acc[c][j] + bias) * scale + shift;
                out[(size_t)g * DIM + col] = y / (1.0f + __expf(-y));
            }
        }
    }
}

extern "C" void kernel_launch(void* const* d_in, const int* in_sizes, int n_in,
                              void* d_out, int out_size, void* d_ws, size_t ws_size,
                              hipStream_t stream) {
    const float* x      = (const float*)d_in[0];
    const float* W      = (const float*)d_in[1];
    const float* bias   = (const float*)d_in[2];
    const float* bn_w   = (const float*)d_in[3];
    const float* bn_b   = (const float*)d_in[4];
    const float* rmean  = (const float*)d_in[5];
    const float* rvar   = (const float*)d_in[6];
    const int*   eidx   = (const int*)d_in[7];
    float*       out    = (float*)d_out;

    int*    ws_i = (int*)d_ws;
    __bf16* Whi  = (__bf16*)((char*)d_ws + OFF_WBF);
    __bf16* Wlo  = Whi + (size_t)NEXP * DIM * DIM;

    zero_meta_k<<<1, 32, 0, stream>>>(ws_i);
    hist_k<<<NTOK / 256, 256, 0, stream>>>(eidx, ws_i);
    build_tiles_k<<<1, 1, 0, stream>>>(ws_i);
    scatter_k<<<NTOK / 256, 256, 0, stream>>>(eidx, ws_i);
    convw_k<<<(NEXP * 32 * 16 * 32) / 256, 256, 0, stream>>>(W, Whi, Wlo);
    moe_gemm_k<<<dim3(MAXTILES, DIM / 64), 256, 0, stream>>>(
        x, bias, bn_w, bn_b, rmean, rvar, ws_i, Whi, Wlo, out);
}
